// ODEClassifier_71399536328921
// MI455X (gfx1250) — matrix operand order repaired and checked by reading, not compiled, not run on hardware
//
#include <hip/hip_runtime.h>

// MI455X / gfx1250 implementation of the Neural-ODE classifier.
// Conv layers are implicit GEMM on v_wmma_f32_16x16x32_f16 (f16 operands,
// f32 accumulate). GroupNorm is fused: stats kernels emit per-(b,c)
// scale/shift; convs apply (v*scale+shift, relu) while staging inputs.
// The ODE time channel (constant t, channel 0) is folded into the epilogue
// via a precomputed border-aware kernel-sum table (t * Ts[cout][y][x]).
//
// ODE conv layout trick: packed-K order is (kh,kw,c) and the LDS input slab
// is [row][col][c], so every WMMA B-fragment is two 8-channel runs in LDS
// (2x ds_load_b128 per lane, no im2col buffer, no K-loop barriers), and
// weight tiles double-buffer per wave via GLOBAL_LOAD_ASYNC_TO_LDS_B128.

typedef __attribute__((ext_vector_type(16))) _Float16 v16h;
typedef __attribute__((ext_vector_type(8)))  _Float16 v8h;
typedef __attribute__((ext_vector_type(8)))  float    v8f;

// ---------------------------------------------------------------------------
// Weight packing / time-channel tables
// ---------------------------------------------------------------------------
__global__ __launch_bounds__(256) void k_pack_f16(const float* __restrict__ s,
                                                  _Float16* __restrict__ d, int n) {
  int i = blockIdx.x * 256 + threadIdx.x;
  if (i < n) d[i] = (_Float16)s[i];
}

// ode_w: [64][65][3][3] f32, channel 0 = time.
// Pack z-channels to [64][576] f16 with K ordered as k = (kh*3+kw)*64 + c.
__global__ __launch_bounds__(256) void k_pack_ode(const float* __restrict__ s,
                                                  _Float16* __restrict__ d) {
  int i = blockIdx.x * 256 + threadIdx.x;
  if (i >= 64 * 576) return;
  int co = i / 576, r = i - co * 576;
  int kk = r >> 6, c = r & 63;  // kk = kh*3+kw
  d[i] = (_Float16)s[co * 585 + (c + 1) * 9 + kk];
}

// Ts[cout][y][x] = sum of time-channel weights whose 3x3 tap lands in-bounds.
__global__ __launch_bounds__(256) void k_timesum(const float* __restrict__ w,
                                                 float* __restrict__ Ts) {
  int i = blockIdx.x * 256 + threadIdx.x;
  if (i >= 64 * 1024) return;
  int co = i >> 10, p = i & 1023, y = p >> 5, x = p & 31;
  float sum = 0.f;
#pragma unroll
  for (int kh = 0; kh < 3; ++kh)
#pragma unroll
    for (int kw = 0; kw < 3; ++kw) {
      int iy = y - 1 + kh, ix = x - 1 + kw;
      if (iy >= 0 && iy < 32 && ix >= 0 && ix < 32) sum += w[co * 585 + kh * 3 + kw];
    }
  Ts[i] = sum;
}

// ---------------------------------------------------------------------------
// GroupNorm stats: per (b,c) over HW, folding gamma/beta -> scale/shift
// ---------------------------------------------------------------------------
template <bool F16>
__global__ __launch_bounds__(256) void k_stats(const void* __restrict__ in_, int HW,
                                               const float* __restrict__ gw,
                                               const float* __restrict__ gb,
                                               float* __restrict__ st) {
  __shared__ float rs[256], rq[256];
  int bc = blockIdx.x, c = bc & 63, tid = threadIdx.x;
  float s = 0.f, q = 0.f;
  if (F16) {
    const _Float16* p = (const _Float16*)in_ + (size_t)bc * HW;
    for (int i = tid; i < HW; i += 256) { float v = (float)p[i]; s += v; q += v * v; }
  } else {
    const float* p = (const float*)in_ + (size_t)bc * HW;
    for (int i = tid; i < HW; i += 256) { float v = p[i]; s += v; q += v * v; }
  }
  rs[tid] = s; rq[tid] = q; __syncthreads();
  for (int o = 128; o > 0; o >>= 1) {
    if (tid < o) { rs[tid] += rs[tid + o]; rq[tid] += rq[tid + o]; }
    __syncthreads();
  }
  if (tid == 0) {
    float inv = 1.0f / (float)HW;
    float m = rs[0] * inv;
    float var = rq[0] * inv - m * m;
    float r = rsqrtf(var + 1e-5f);
    float a = r * gw[c];
    st[bc * 2 + 0] = a;
    st[bc * 2 + 1] = gb[c] - m * a;
  }
}

// ---------------------------------------------------------------------------
// Downsampler stage 1: conv(1->64,3x3,VALID) + GN + ReLU, fused two-pass
// (conv recomputed after the stats reduction -- 9 MACs/px, cheaper than
// materializing a 256MB intermediate). Output f16 [64][64][128][128].
// ---------------------------------------------------------------------------
__global__ __launch_bounds__(256) void k_ds1(const float* __restrict__ x,
                                             const float* __restrict__ w1,
                                             const float* __restrict__ b1,
                                             const float* __restrict__ gw,
                                             const float* __restrict__ gb,
                                             _Float16* __restrict__ act1) {
  __shared__ float rs[256], rq[256];
  __shared__ float sA, sB;
  int bc = blockIdx.x, b = bc >> 6, c = bc & 63, tid = threadIdx.x;
  float w[9];
#pragma unroll
  for (int j = 0; j < 9; ++j) w[j] = w1[c * 9 + j];
  float bias = b1[c];
  const float* xb = x + (size_t)b * 130 * 130;

  float s = 0.f, q = 0.f;
  for (int p = tid; p < 16384; p += 256) {
    int y = p >> 7, xx = p & 127;
    float v = bias;
#pragma unroll
    for (int kh = 0; kh < 3; ++kh)
#pragma unroll
      for (int kw = 0; kw < 3; ++kw)
        v += xb[(y + kh) * 130 + xx + kw] * w[kh * 3 + kw];
    s += v; q += v * v;
  }
  rs[tid] = s; rq[tid] = q; __syncthreads();
  for (int o = 128; o > 0; o >>= 1) {
    if (tid < o) { rs[tid] += rs[tid + o]; rq[tid] += rq[tid + o]; }
    __syncthreads();
  }
  if (tid == 0) {
    float m = rs[0] * (1.f / 16384.f);
    float var = rq[0] * (1.f / 16384.f) - m * m;
    float r = rsqrtf(var + 1e-5f);
    float a = r * gw[c];
    sA = a; sB = gb[c] - m * a;
  }
  __syncthreads();
  float a = sA, sh = sB;
  _Float16* dst = act1 + (size_t)bc * 16384;
  for (int p = tid; p < 16384; p += 256) {
    int y = p >> 7, xx = p & 127;
    float v = bias;
#pragma unroll
    for (int kh = 0; kh < 3; ++kh)
#pragma unroll
      for (int kw = 0; kw < 3; ++kw)
        v += xb[(y + kh) * 130 + xx + kw] * w[kh * 3 + kw];
    v = v * a + sh;
    dst[p] = (_Float16)(v > 0.f ? v : 0.f);
  }
}

// ---------------------------------------------------------------------------
// Generic WMMA implicit-GEMM conv (used for downsampler conv2/conv3).
// M=Cout=64, N tile 128/block, K=64*KH*KW in chunks of 32 through LDS.
// ---------------------------------------------------------------------------
template <int KH, int KW, int STR, int PAD, int HIN, int WIN, int HOUT, int WOUT,
          bool IN_F16, bool GN, bool OUT_F16>
__global__ __launch_bounds__(256) void k_conv_wmma(
    const void* __restrict__ in_, const _Float16* __restrict__ Wp,
    const float* __restrict__ bias, const float* __restrict__ stats,
    void* __restrict__ out_) {
  constexpr int KHW = KH * KW;
  constexpr int K = 64 * KHW;
  constexpr int CH = 32;  // K chunk

  __shared__ __align__(16) _Float16 As[64 * CH];    // [M][k] weights
  __shared__ __align__(16) _Float16 Bs[128 * CH];   // [n][k] im2col patches

  const int b = blockIdx.y;
  const int p0 = blockIdx.x * 128;
  const int tid = threadIdx.x;
  const int wave = tid >> 5, lane = tid & 31;
  const int wm = wave & 3, wn = wave >> 2;
  const int lgrp = (lane < 16) ? 0 : 1;
  const int l15 = lane & 15;

  const v8f vzero = {0.f, 0.f, 0.f, 0.f, 0.f, 0.f, 0.f, 0.f};
  v8f acc[4] = {vzero, vzero, vzero, vzero};

  const _Float16* inH = (const _Float16*)in_;
  const float* inF = (const float*)in_;

  for (int k0 = 0; k0 < K; k0 += CH) {
    // ---- stage A (weights): 256 threads x v8h covers 64x32 halfs ----
    {
      int m = tid >> 2;
      int ko = (tid & 3) * 8;
      const _Float16* src = Wp + (size_t)m * K + k0 + ko;
      v8h w = *(const v8h*)src;
      *(v8h*)(&As[m * CH + ko]) = w;
      if (k0 + CH < K) __builtin_prefetch(src + CH, 0, 0);  // global_prefetch_b8
    }
    // ---- stage B (im2col + fused GN/ReLU): thread -> (n, 16 k's) ----
    {
      int n = tid >> 1;
      int kb = (tid & 1) * 16;
      int p = p0 + n;
      int y = p / WOUT;
      int x = p - y * WOUT;
      v8h t0, t1;
#pragma unroll
      for (int j = 0; j < 16; ++j) {
        int k = k0 + kb + j;
        int c = k / KHW;
        int r = k - c * KHW;
        int kh = r / KW;
        int kw = r - kh * KW;
        int iy = y * STR - PAD + kh;
        int ix = x * STR - PAD + kw;
        float v = 0.f;
        if (iy >= 0 && iy < HIN && ix >= 0 && ix < WIN) {
          size_t idx = (((size_t)b * 64 + c) * HIN + iy) * WIN + ix;
          v = IN_F16 ? (float)inH[idx] : inF[idx];
          if (GN) {
            int bc = b * 64 + c;
            v = v * stats[bc * 2] + stats[bc * 2 + 1];
            v = v > 0.f ? v : 0.f;
          }
        }
        if (j < 8) t0[j] = (_Float16)v; else t1[j - 8] = (_Float16)v;
      }
      v8h* dst = (v8h*)(&Bs[n * CH + kb]);
      dst[0] = t0;
      dst[1] = t1;
    }
    __syncthreads();
    // ---- WMMA: fragment layouts per CDNA5 ISA 7.12.2 ----
    {
      int m = wm * 16 + l15;
      int o0 = lgrp * 8;  // lanes 0-15: K{0..7,16..23}; 16-31: K{8..15,24..31}
      v8h alo = *(const v8h*)(&As[m * CH + o0]);
      v8h ahi = *(const v8h*)(&As[m * CH + o0 + 16]);
      v16h a = __builtin_shufflevector(alo, ahi, 0, 1, 2, 3, 4, 5, 6, 7, 8, 9,
                                       10, 11, 12, 13, 14, 15);
#pragma unroll
      for (int j = 0; j < 4; ++j) {
        int n = wn * 64 + j * 16 + l15;
        int bo = lgrp * 8;  // lanes 0-15: K{0..7,16..23}; 16-31: K{8..15,24..31}
        v8h blo = *(const v8h*)(&Bs[n * CH + bo]);
        v8h bhi = *(const v8h*)(&Bs[n * CH + bo + 16]);
        v16h bb = __builtin_shufflevector(blo, bhi, 0, 1, 2, 3, 4, 5, 6, 7, 8,
                                          9, 10, 11, 12, 13, 14, 15);
        acc[j] = __builtin_amdgcn_wmma_f32_16x16x32_f16(
            false, a, false, bb, (short)0, acc[j], false, false);
      }
    }
    __syncthreads();
  }

  // ---- epilogue: bias, store f32/f16 ----
  _Float16* outH = (_Float16*)out_;
  float* outF = (float*)out_;
#pragma unroll
  for (int j = 0; j < 4; ++j) {
#pragma unroll
    for (int r = 0; r < 8; ++r) {
      int m = wm * 16 + r + lgrp * 8;       // C layout: lanes 16-31 -> M+8
      int p = p0 + wn * 64 + j * 16 + l15;  // N = lane%16
      int y = p / WOUT, x = p - y * WOUT;
      float v = acc[j][r] + bias[m];
      size_t oi = (((size_t)b * 64 + m) * HOUT + y) * WOUT + x;
      if (OUT_F16) outH[oi] = (_Float16)v; else outF[oi] = v;
    }
  }
}

// ---------------------------------------------------------------------------
// ODE conv (3x3, s1, p1, 32x32, Cin=Cout=64): the 80-instance hot loop.
// Input slab [6 rows][34 cols][64 ch] (GN+ReLU'd f16, halo zeroed) staged
// once in LDS with channels innermost: a 32-wide K chunk has fixed (kh,kw)
// and contiguous channels, so every B fragment is 2x ds_load_b128 straight
// from the slab -- no im2col buffer, no barriers in the K loop.
// Weights double-buffer per wave via GLOBAL_LOAD_ASYNC_TO_LDS_B128.
// ---------------------------------------------------------------------------
__global__ __launch_bounds__(256) void k_conv_ode(
    const float* __restrict__ in, const _Float16* __restrict__ Wp,
    const float* __restrict__ bias, const float* __restrict__ stats,
    const float* __restrict__ Ts, float t, float* __restrict__ out) {
  constexpr int K = 576, CH = 32;

  __shared__ __align__(16) _Float16 Raw[6 * 34 * 64];   // 25.5 KB input slab
  __shared__ __align__(16) _Float16 Aw[8][2][16 * 32];  // 16 KB per-wave dbl-buf

  const int b = blockIdx.y;
  const int r0 = blockIdx.x * 4;   // 4 output rows per block
  const int p0 = blockIdx.x * 128;
  const int tid = threadIdx.x;
  const int wave = tid >> 5, lane = tid & 31;
  const int wm = wave & 3, wn = wave >> 2;
  const int lgrp = lane >> 4;
  const int l15 = lane & 15;

  // ---- kick off chunk-0 weight prefetch (per-wave async DMA to LDS) ----
  const int arow = l15;             // local A row; lanes 0-15 copy k0..15,
  const int akoff = lgrp * 16;      // lanes 16-31 copy k16..31 of same row
  const _Float16* wsrc = Wp + (size_t)(wm * 16 + arow) * K + akoff;
  _Float16* abuf0 = &Aw[wave][0][arow * 32 + akoff];
  _Float16* abuf1 = &Aw[wave][1][arow * 32 + akoff];
  unsigned adst0 = (unsigned)(size_t)abuf0;  // low 32b = LDS offset
  unsigned adst1 = (unsigned)(size_t)abuf1;
  asm volatile("global_load_async_to_lds_b128 %0, %1, off"
               :: "v"(adst0), "v"(wsrc) : "memory");
  asm volatile("global_load_async_to_lds_b128 %0, %1, off offset:16"
               :: "v"(adst0), "v"(wsrc) : "memory");

  // ---- stage input slab: GN+ReLU fused, f16, transposed to c-innermost ----
  {
    const v8h hz = {(_Float16)0.f, (_Float16)0.f, (_Float16)0.f, (_Float16)0.f,
                    (_Float16)0.f, (_Float16)0.f, (_Float16)0.f, (_Float16)0.f};
    for (int idx = tid; idx < 96; idx += 256) {  // halo columns 0 and 33
      int rr = idx >> 4, rest = idx & 15;
      int col = (rest & 8) ? 33 : 0;
      int g = rest & 7;
      *(v8h*)(&Raw[((rr * 34) + col) * 64 + g * 8]) = hz;
    }
  }
  for (int idx = tid; idx < 3072; idx += 256) {  // (c, row, 4-col group)
    int j = idx >> 3, cg = idx & 7;
    int c = j / 6, rr = j - c * 6;
    int iy = r0 - 1 + rr;
    int col0 = cg * 4;
    float4 v = {0.f, 0.f, 0.f, 0.f};
    float sc = 0.f, sh = 0.f;
    if (iy >= 0 && iy < 32) {
      v = *(const float4*)(in + (((size_t)b * 64 + c) * 32 + iy) * 32 + col0);
      int bc = b * 64 + c;
      sc = stats[bc * 2];
      sh = stats[bc * 2 + 1];
    }
    int base = (rr * 34 + col0 + 1) * 64 + c;  // col index = ix + 1
    Raw[base + 0 * 64] = (_Float16)fmaxf(fmaf(v.x, sc, sh), 0.f);
    Raw[base + 1 * 64] = (_Float16)fmaxf(fmaf(v.y, sc, sh), 0.f);
    Raw[base + 2 * 64] = (_Float16)fmaxf(fmaf(v.z, sc, sh), 0.f);
    Raw[base + 3 * 64] = (_Float16)fmaxf(fmaf(v.w, sc, sh), 0.f);
  }
  __syncthreads();  // slab is read-only below; the only block-wide barrier

  const v8f vzero = {0.f, 0.f, 0.f, 0.f, 0.f, 0.f, 0.f, 0.f};
  v8f acc[4] = {vzero, vzero, vzero, vzero};

  for (int k0 = 0, it = 0; k0 < K; k0 += CH, ++it) {
    const int cur = it & 1;
    // prefetch next chunk into the other buffer, then drain current pair
    if (k0 + CH < K) {
      const _Float16* sn = wsrc + k0 + CH;
      unsigned dn = cur ? adst0 : adst1;
      asm volatile("global_load_async_to_lds_b128 %0, %1, off"
                   :: "v"(dn), "v"(sn) : "memory");
      asm volatile("global_load_async_to_lds_b128 %0, %1, off offset:16"
                   :: "v"(dn), "v"(sn) : "memory");
      asm volatile("s_wait_asynccnt 0x2" ::: "memory");
    } else {
      asm volatile("s_wait_asynccnt 0x0" ::: "memory");
    }
    // A fragment from this wave's private buffer
    const _Float16* Ab = &Aw[wave][cur][0];
    v8h alo = *(const v8h*)(Ab + arow * 32 + lgrp * 8);
    v8h ahi = *(const v8h*)(Ab + arow * 32 + lgrp * 8 + 16);
    v16h a = __builtin_shufflevector(alo, ahi, 0, 1, 2, 3, 4, 5, 6, 7, 8, 9,
                                     10, 11, 12, 13, 14, 15);
    // B fragments straight from the slab (contiguous channels)
    const int kk = k0 >> 6;       // kh*3+kw
    const int c0 = k0 & 63;       // 0 or 32
    const int kh = kk / 3, kw = kk - kh * 3;
#pragma unroll
    for (int j = 0; j < 4; ++j) {
      int px = wn * 64 + j * 16 + l15;  // pixel within the 128-tile
      int ry = px >> 5;                 // row within the 4-row tile
      int x = px & 31;
      int base = ((ry + kh) * 34 + (x + kw)) * 64 + c0 + lgrp * 8;
      v8h blo = *(const v8h*)(&Raw[base]);
      v8h bhi = *(const v8h*)(&Raw[base + 16]);
      v16h bb = __builtin_shufflevector(blo, bhi, 0, 1, 2, 3, 4, 5, 6, 7, 8, 9,
                                        10, 11, 12, 13, 14, 15);
      acc[j] = __builtin_amdgcn_wmma_f32_16x16x32_f16(
          false, a, false, bb, (short)0, acc[j], false, false);
    }
  }

  // ---- epilogue: bias + t * time-kernel sum, f32 out ----
#pragma unroll
  for (int j = 0; j < 4; ++j) {
#pragma unroll
    for (int r = 0; r < 8; ++r) {
      int m = wm * 16 + r + lgrp * 8;
      int p = p0 + wn * 64 + j * 16 + l15;
      int y = p >> 5, x = p & 31;
      float v = acc[j][r] + bias[m] + t * Ts[m * 1024 + y * 32 + x];
      out[(((size_t)b * 64 + m) * 32 + y) * 32 + x] = v;
    }
  }
}

// ---------------------------------------------------------------------------
// RK4 elementwise combine
// ---------------------------------------------------------------------------
__global__ __launch_bounds__(256) void k_rk_stage(const float* __restrict__ z,
                                                  const float* __restrict__ k,
                                                  float* __restrict__ acc,
                                                  float* __restrict__ zt,
                                                  float a, float s, int first, int n) {
  int i = blockIdx.x * 256 + threadIdx.x;
  if (i >= n) return;
  float kv = k[i];
  acc[i] = first ? kv : fmaf(s, kv, acc[i]);
  zt[i] = fmaf(a, kv, z[i]);
}

__global__ __launch_bounds__(256) void k_rk_final(float* __restrict__ z,
                                                  const float* __restrict__ k,
                                                  const float* __restrict__ acc,
                                                  float c, int n) {
  int i = blockIdx.x * 256 + threadIdx.x;
  if (i >= n) return;
  z[i] = fmaf(c, acc[i] + k[i], z[i]);
}

// ---------------------------------------------------------------------------
// Head: GN+ReLU (via scale/shift) + GAP + Linear(64->10). One block per batch.
// ---------------------------------------------------------------------------
__global__ __launch_bounds__(256) void k_head(const float* __restrict__ z,
                                              const float* __restrict__ stats,
                                              const float* __restrict__ W,
                                              const float* __restrict__ bias,
                                              float* __restrict__ out) {
  __shared__ float partial[256];
  __shared__ float feat[64];
  int b = blockIdx.x, tid = threadIdx.x;
  int c = tid >> 2, part = tid & 3;
  float sc = stats[(b * 64 + c) * 2], sh = stats[(b * 64 + c) * 2 + 1];
  const float* zp = z + ((size_t)b * 64 + c) * 1024;
  float s = 0.f;
  for (int p = part * 256; p < part * 256 + 256; ++p) {
    float v = fmaf(zp[p], sc, sh);
    s += v > 0.f ? v : 0.f;
  }
  partial[tid] = s;
  __syncthreads();
  if (part == 0)
    feat[c] = (partial[tid] + partial[tid + 1] + partial[tid + 2] + partial[tid + 3]) *
              (1.0f / 1024.0f);
  __syncthreads();
  if (tid < 10) {
    float acc = bias[tid];
    for (int c2 = 0; c2 < 64; ++c2) acc += feat[c2] * W[tid * 64 + c2];
    out[b * 10 + tid] = acc;
  }
}

// ---------------------------------------------------------------------------
// Host orchestration
// ---------------------------------------------------------------------------
extern "C" void kernel_launch(void* const* d_in, const int* in_sizes, int n_in,
                              void* d_out, int out_size, void* d_ws, size_t ws_size,
                              hipStream_t stream) {
  (void)in_sizes; (void)n_in; (void)out_size; (void)ws_size;

  const float* x       = (const float*)d_in[0];
  const float* ds_w1   = (const float*)d_in[1];
  const float* ds_b1   = (const float*)d_in[2];
  const float* ds_n1w  = (const float*)d_in[3];
  const float* ds_n1b  = (const float*)d_in[4];
  const float* ds_w2   = (const float*)d_in[5];
  const float* ds_b2   = (const float*)d_in[6];
  const float* ds_n2w  = (const float*)d_in[7];
  const float* ds_n2b  = (const float*)d_in[8];
  const float* ds_w3   = (const float*)d_in[9];
  const float* ds_b3   = (const float*)d_in[10];
  const float* ode_w1  = (const float*)d_in[11];
  const float* ode_b1  = (const float*)d_in[12];
  const float* ode_n1w = (const float*)d_in[13];
  const float* ode_n1b = (const float*)d_in[14];
  const float* ode_w2  = (const float*)d_in[15];
  const float* ode_b2  = (const float*)d_in[16];
  const float* ode_n2w = (const float*)d_in[17];
  const float* ode_n2b = (const float*)d_in[18];
  const float* head_nw = (const float*)d_in[19];
  const float* head_nb = (const float*)d_in[20];
  const float* head_W  = (const float*)d_in[21];
  const float* head_b  = (const float*)d_in[22];

  // Workspace arena (peak ~169 MB). act1 (dead after conv2) overlaps the
  // five 16.8MB f32 ODE buffers.
  char* ws = (char*)d_ws;
  constexpr size_t SZ_Z = (size_t)64 * 64 * 32 * 32 * 4;          // 16.8 MB
  float* z    = (float*)(ws + 0 * SZ_Z);
  float* accb = (float*)(ws + 1 * SZ_Z);
  float* zt   = (float*)(ws + 2 * SZ_Z);
  float* g1   = (float*)(ws + 3 * SZ_Z);
  float* kout = (float*)(ws + 4 * SZ_Z);
  constexpr size_t SZ_ACT1 = (size_t)64 * 64 * 128 * 128 * 2;     // 134 MB
  _Float16* act1 = (_Float16*)(ws + 0);
  constexpr size_t OFF_C2 = SZ_ACT1;
  _Float16* c2 = (_Float16*)(ws + OFF_C2);                        // 33.5 MB
  size_t off = OFF_C2 + (size_t)64 * 64 * 64 * 64 * 2;
  float* stats  = (float*)(ws + off);  off += 4096 * 2 * 4;
  _Float16* Wp2 = (_Float16*)(ws + off); off += (size_t)64 * 1024 * 2;
  _Float16* Wp3 = (_Float16*)(ws + off); off += (size_t)64 * 1024 * 2;
  _Float16* Wo1 = (_Float16*)(ws + off); off += (size_t)64 * 576 * 2;
  _Float16* Wo2 = (_Float16*)(ws + off); off += (size_t)64 * 576 * 2;
  float* Ts1    = (float*)(ws + off);  off += (size_t)64 * 1024 * 4;
  float* Ts2    = (float*)(ws + off);  off += (size_t)64 * 1024 * 4;

  // Weight prep
  k_pack_f16<<<dim3(256), 256, 0, stream>>>(ds_w2, Wp2, 65536);
  k_pack_f16<<<dim3(256), 256, 0, stream>>>(ds_w3, Wp3, 65536);
  k_pack_ode<<<dim3(144), 256, 0, stream>>>(ode_w1, Wo1);
  k_pack_ode<<<dim3(144), 256, 0, stream>>>(ode_w2, Wo2);
  k_timesum<<<dim3(256), 256, 0, stream>>>(ode_w1, Ts1);
  k_timesum<<<dim3(256), 256, 0, stream>>>(ode_w2, Ts2);

  // Downsampler
  k_ds1<<<dim3(4096), 256, 0, stream>>>(x, ds_w1, ds_b1, ds_n1w, ds_n1b, act1);

  k_conv_wmma<4, 4, 2, 1, 128, 128, 64, 64, true, false, true>
      <<<dim3(32, 64), 256, 0, stream>>>(act1, Wp2, ds_b2, nullptr, c2);
  k_stats<true><<<dim3(4096), 256, 0, stream>>>(c2, 4096, ds_n2w, ds_n2b, stats);
  k_conv_wmma<4, 4, 2, 1, 64, 64, 32, 32, true, true, false>
      <<<dim3(8, 64), 256, 0, stream>>>(c2, Wp3, ds_b3, stats, z);

  // ODE RHS: relu(gn1(z)) -> conv1(+t) -> relu(gn2) -> conv2(+t)
  auto rhs = [&](float t, const float* zin, float* out) {
    k_stats<false><<<dim3(4096), 256, 0, stream>>>(zin, 1024, ode_n1w, ode_n1b, stats);
    k_conv_ode<<<dim3(8, 64), 256, 0, stream>>>(zin, Wo1, ode_b1, stats, Ts1, t, g1);
    k_stats<false><<<dim3(4096), 256, 0, stream>>>(g1, 1024, ode_n2w, ode_n2b, stats);
    k_conv_ode<<<dim3(8, 64), 256, 0, stream>>>(g1, Wo2, ode_b2, stats, Ts2, t, out);
  };

  const float dt = 0.1f;
  const int NEL = 64 * 64 * 32 * 32;
  const int NB = NEL / 256;
  for (int i = 0; i < 10; ++i) {
    float t = dt * (float)i;
    rhs(t, z, kout);  // k1
    k_rk_stage<<<dim3(NB), 256, 0, stream>>>(z, kout, accb, zt, 0.5f * dt, 0.f, 1, NEL);
    rhs(t + 0.5f * dt, zt, kout);  // k2
    k_rk_stage<<<dim3(NB), 256, 0, stream>>>(z, kout, accb, zt, 0.5f * dt, 2.f, 0, NEL);
    rhs(t + 0.5f * dt, zt, kout);  // k3
    k_rk_stage<<<dim3(NB), 256, 0, stream>>>(z, kout, accb, zt, dt, 2.f, 0, NEL);
    rhs(t + dt, zt, kout);  // k4
    k_rk_final<<<dim3(NB), 256, 0, stream>>>(z, kout, accb, dt / 6.f, NEL);
  }

  // Head
  k_stats<false><<<dim3(4096), 256, 0, stream>>>(z, 1024, head_nw, head_nb, stats);
  k_head<<<dim3(64), 256, 0, stream>>>(z, stats, head_W, head_b, (float*)d_out);
}
